// ContextualEncoder_90683939488368
// MI455X (gfx1250) — matrix operand order repaired and checked by reading, not compiled, not run on hardware
//
#include <hip/hip_runtime.h>
#include <cstdint>
#include <cstddef>

// ---------------------------------------------------------------------------
// ContextualEncoder (5x fused GRU cells) for MI455X / gfx1250.
// Memory-bound (~1 GB min traffic @ 23.3 TB/s). bf16 WMMA, f32 accumulate,
// fused gate math. Weight tiles double-buffered in LDS via async-to-LDS
// loads (ASYNCcnt) so all 8 waves share one copy of the B operand.
// ---------------------------------------------------------------------------

#define HID   1024
#define DIN   1088
#define NB    16384
#define FEAT  64
#define OBSLD 192   // (TIME_STEPS=3) * FEAT

typedef __attribute__((ext_vector_type(16))) __bf16 v16bf;
typedef __attribute__((ext_vector_type(8)))  float  v8f;
typedef __attribute__((ext_vector_type(4)))  int    v4i;

union FragU { v16bf v; uint4 q[2]; };

// 16-byte cooperative copy into LDS: async (gfx1250 path) or sync fallback.
#if __has_builtin(__builtin_amdgcn_global_load_async_to_lds_b128)
#define HAVE_ASYNC_LDS 1
#else
#define HAVE_ASYNC_LDS 0
#endif

__device__ __forceinline__ void cp16_to_lds(uint4* l, const __bf16* g) {
#if HAVE_ASYNC_LDS
    __builtin_amdgcn_global_load_async_to_lds_b128(
        (__attribute__((address_space(1))) v4i*)g,
        (__attribute__((address_space(3))) v4i*)l, 0, 0);
#else
    *l = *(const uint4*)g;
#endif
}

__device__ __forceinline__ void wait_async_lds() {
#if HAVE_ASYNC_LDS
#if __has_builtin(__builtin_amdgcn_s_wait_asynccnt)
    __builtin_amdgcn_s_wait_asynccnt(0);
#else
    asm volatile("s_wait_asynccnt 0x0" ::: "memory");
#endif
#endif
}

// A-fragment (16x32 bf16, row-major source):
// lane l: row=l%16, kh=l/16; a[0..7]=X[row][k0+8*kh+i], a[8..15]=X[row][k0+16+8*kh+i]
__device__ __forceinline__ v16bf load_afrag(const __bf16* base, int ld,
                                            int row0, int k0, int lane) {
    const int r  = lane & 15;
    const int kh = (lane >> 4) & 1;
    const __bf16* p = base + (size_t)(row0 + r) * (size_t)ld + (size_t)(k0 + kh * 8);
    FragU f;
    f.q[0] = *(const uint4*)(p);
    f.q[1] = *(const uint4*)(p + 16);
    return f.v;
}

// B-fragment (32x16 bf16) from the LDS-staged weight tile.
// Tile: 96 rows (3 gates x 32 cols) x 32 K, row pitch 5 uint4 (80B, padded).
// lane l: c=l%16, kh=l/16; b[0..7] = Wtile[g*32 + ct*16 + c][kh*8 + i], b[8..15] = Wtile[...][16 + kh*8 + i]
__device__ __forceinline__ v16bf load_bfrag_lds(const uint4* tile,
                                                int g, int ct, int lane) {
    const int c  = lane & 15;
    const int kh = (lane >> 4) & 1;
    const uint4* p = tile + (g * 32 + ct * 16 + c) * 5 + kh;
    FragU f;
    f.q[0] = p[0];
    f.q[1] = p[2];
    return f.v;
}

// Fused GRU cell: h' = (1-z)*n + z*h, r/z = sigmoid(gi+gh), n = tanh(i_n + r*h_n)
// x = concat(xA[wA], xB[wB]) handled as K-phases with a column offset in Wih;
// phase 2 is the gh = h @ Whh^T GEMM. One software-pipelined K loop covers all.
// Workgroup: 8 waves, 128 rows x 32 cols. Wave: 16 rows x 32 cols, 12 f32 accs.
__global__ __launch_bounds__(256)
void gru_cell_wmma(const __bf16* __restrict__ xA, int ldxA, int wA,
                   const __bf16* __restrict__ xB, int ldxB, int wB,
                   const __bf16* __restrict__ hin,
                   const __bf16* __restrict__ Wih,   // 3072 x 1088 bf16
                   const __bf16* __restrict__ Whh,   // 3072 x 1024 bf16
                   const float*  __restrict__ bih,   // 3072 f32
                   const float*  __restrict__ bhh,   // 3072 f32
                   __bf16* __restrict__ hout_bf,
                   float*  __restrict__ hout_f32)    // may be nullptr
{
    const int tid  = threadIdx.x;
    const int lane = tid & 31;
    const int wave = tid >> 5;
    const int rowBase = blockIdx.y * 128 + wave * 16;
    const int colBase = blockIdx.x * 32;

    __shared__ uint4 smem[2][96 * 5];   // double-buffered weight tile, 15 KB

    v8f accI[6];   // [gate(r,z,n)*2 + colsub]
    v8f accH[6];
#pragma unroll
    for (int i = 0; i < 6; ++i) {
        accI[i] = (v8f){0.f,0.f,0.f,0.f,0.f,0.f,0.f,0.f};
        accH[i] = (v8f){0.f,0.f,0.f,0.f,0.f,0.f,0.f,0.f};
    }

    const int T0 = wA >> 5;             // phase 0 steps (xA . Wih[:, 0:wA])
    const int T1 = T0 + (wB >> 5);      // phase 1 steps (xB . Wih[:, wA:])
    const int T  = T1 + (HID >> 5);     // phase 2 steps (h  . Whh)

    // Cooperatively stage the 96x32 weight tile for step s into smem[s&1].
    auto issue_tile = [&](int s) {
        const int ph = (s < T0) ? 0 : ((s < T1) ? 1 : 2);
        const int klocal = (s - ((ph == 0) ? 0 : ((ph == 1) ? T0 : T1))) << 5;
        const __bf16* W;
        int ldw, kcol;
        if (ph == 2) { W = Whh; ldw = HID; kcol = klocal; }
        else         { W = Wih; ldw = DIN; kcol = ((ph == 1) ? wA : 0) + klocal; }
        uint4* buf = smem[s & 1];
        // 96 rows x 4 chunks of 16B = 384 chunks; 256 threads.
#pragma unroll 1
        for (int ch = tid; ch < 384; ch += 256) {
            const int row = ch >> 2, part = ch & 3;
            const int g = row >> 5, c = row & 31;
            const __bf16* gp = W + (size_t)(g * HID + colBase + c) * (size_t)ldw
                                 + (size_t)(kcol + part * 8);
            cp16_to_lds(buf + row * 5 + part, gp);
        }
    };

    issue_tile(0);
    wait_async_lds();
    __syncthreads();

#pragma unroll 1
    for (int s = 0; s < T; ++s) {
        if (s + 1 < T) issue_tile(s + 1);   // prefetch next tile (other buffer)

        const int ph = (s < T0) ? 0 : ((s < T1) ? 1 : 2);
        const int klocal = (s - ((ph == 0) ? 0 : ((ph == 1) ? T0 : T1))) << 5;
        const __bf16* X;
        int ldx;
        if      (ph == 0) { X = xA;  ldx = ldxA; }
        else if (ph == 1) { X = xB;  ldx = ldxB; }
        else              { X = hin; ldx = HID;  }

        const v16bf a = load_afrag(X, ldx, rowBase, klocal, lane);
        const uint4* tile = smem[s & 1];
        if (ph < 2) {
#pragma unroll
            for (int g = 0; g < 3; ++g)
#pragma unroll
                for (int ct = 0; ct < 2; ++ct) {
                    v16bf b = load_bfrag_lds(tile, g, ct, lane);
                    accI[g * 2 + ct] = __builtin_amdgcn_wmma_f32_16x16x32_bf16(
                        false, a, false, b, (short)0, accI[g * 2 + ct], false, false);
                }
        } else {
#pragma unroll
            for (int g = 0; g < 3; ++g)
#pragma unroll
                for (int ct = 0; ct < 2; ++ct) {
                    v16bf b = load_bfrag_lds(tile, g, ct, lane);
                    accH[g * 2 + ct] = __builtin_amdgcn_wmma_f32_16x16x32_bf16(
                        false, a, false, b, (short)0, accH[g * 2 + ct], false, false);
                }
        }

        wait_async_lds();   // next tile landed; also fences before reuse
        __syncthreads();
    }

    // Gate math. C/D layout: lane l -> col = l%16; VGPR v -> row = v + 8*(l/16).
    const int n_   = lane & 15;
    const int rhLO = (lane >> 4) & 1;
#pragma unroll
    for (int ct = 0; ct < 2; ++ct) {
        const int col = colBase + ct * 16 + n_;
        const float bir = bih[col],           bhr = bhh[col];
        const float biz = bih[col + HID],     bhz = bhh[col + HID];
        const float bin = bih[col + 2 * HID], bhn = bhh[col + 2 * HID];
#pragma unroll
        for (int v = 0; v < 8; ++v) {
            const int row = rowBase + v + 8 * rhLO;
            const float ir = accI[0 * 2 + ct][v] + bir;
            const float iz = accI[1 * 2 + ct][v] + biz;
            const float in = accI[2 * 2 + ct][v] + bin;
            const float hr = accH[0 * 2 + ct][v] + bhr;
            const float hz = accH[1 * 2 + ct][v] + bhz;
            const float hn = accH[2 * 2 + ct][v] + bhn;
            const float r  = 1.f / (1.f + __expf(-(ir + hr)));
            const float z  = 1.f / (1.f + __expf(-(iz + hz)));
            const float n  = tanhf(in + r * hn);
            const float hp = (float)hin[(size_t)row * HID + col];
            const float hv = (1.f - z) * n + z * hp;
            hout_bf[(size_t)row * HID + col] = (__bf16)hv;
            if (hout_f32) hout_f32[(size_t)row * HID + col] = hv;
        }
    }
}

__global__ void cvt_f32_bf16(const float* __restrict__ s, __bf16* __restrict__ d,
                             size_t n) {
    size_t i = (size_t)blockIdx.x * blockDim.x + threadIdx.x;
    size_t stride = (size_t)gridDim.x * blockDim.x;
    for (; i < n; i += stride) d[i] = (__bf16)s[i];
}

// Resolve the embedding gather once into a dense (NB x HID) bf16 operand.
__global__ void gather_emb_bf16(const int* __restrict__ ids,
                                const float* __restrict__ tab,
                                __bf16* __restrict__ out) {
    size_t i = (size_t)blockIdx.x * blockDim.x + threadIdx.x;
    size_t total = (size_t)NB * HID;
    size_t stride = (size_t)gridDim.x * blockDim.x;
    for (; i < total; i += stride) {
        size_t nrow = i >> 10;          // / HID
        size_t k    = i & (HID - 1);
        out[i] = (__bf16)tab[(size_t)ids[nrow] * HID + k];
    }
}

extern "C" void kernel_launch(void* const* d_in, const int* in_sizes, int n_in,
                              void* d_out, int out_size, void* d_ws, size_t ws_size,
                              hipStream_t stream) {
    const float* buoy_obs = (const float*)d_in[0];  // (NB, 3, 64)
    const int*   buoy_ids = (const int*)  d_in[1];  // (NB,)
    const float* emb_tab  = (const float*)d_in[2];  // (100, 1024)
    const float* w_ih     = (const float*)d_in[3];  // (3072, 1088)
    const float* w_hh     = (const float*)d_in[4];  // (3072, 1024)
    const float* b_ih     = (const float*)d_in[5];  // (3072,)
    const float* b_hh     = (const float*)d_in[6];  // (3072,)
    float* out = (float*)d_out;                     // (1, NB, 1024)

    // Workspace carve-up (256B aligned), ~154 MB total.
    char* p = (char*)d_ws;
    auto carve = [&](size_t bytes) {
        void* r = (void*)p;
        p += (bytes + 255) & ~(size_t)255;
        return r;
    };
    const size_t WIH_E = (size_t)3 * HID * DIN;
    const size_t WHH_E = (size_t)3 * HID * HID;
    const size_t OBS_E = (size_t)NB * OBSLD;
    const size_t H_E   = (size_t)NB * HID;

    __bf16* Wih_b = (__bf16*)carve(WIH_E * 2);
    __bf16* Whh_b = (__bf16*)carve(WHH_E * 2);
    __bf16* obs_b = (__bf16*)carve(OBS_E * 2);
    __bf16* emb_b = (__bf16*)carve(H_E * 2);
    __bf16* h0    = (__bf16*)carve(H_E * 2);
    __bf16* h1    = (__bf16*)carve(H_E * 2);
    __bf16* out0  = (__bf16*)carve(H_E * 2);   // step-0 hidden, reused in pass 4

    const int CB = 256;
    // One-time precision / gather prep
    cvt_f32_bf16<<<2048, CB, 0, stream>>>(w_ih, Wih_b, WIH_E);
    cvt_f32_bf16<<<2048, CB, 0, stream>>>(w_hh, Whh_b, WHH_E);
    cvt_f32_bf16<<<2048, CB, 0, stream>>>(buoy_obs, obs_b, OBS_E);
    gather_emb_bf16<<<4096, CB, 0, stream>>>(buoy_ids, emb_tab, emb_b);
    (void)hipMemsetAsync(h0, 0, H_E * 2, stream);   // h initial state = 0

    dim3 grid(HID / 32, NB / 128);             // 32 x 128 workgroups
    dim3 block(256);                           // 8 waves of 32

    // t=0: x=[obs0 | emb], h:0 -> out0
    gru_cell_wmma<<<grid, block, 0, stream>>>(obs_b + 0,   OBSLD, FEAT,
                                              emb_b,       HID,   HID,
                                              h0, Wih_b, Whh_b, b_ih, b_hh,
                                              out0, nullptr);
    // t=1: x=[obs1 | emb], h:out0 -> h1
    gru_cell_wmma<<<grid, block, 0, stream>>>(obs_b + 64,  OBSLD, FEAT,
                                              emb_b,       HID,   HID,
                                              out0, Wih_b, Whh_b, b_ih, b_hh,
                                              h1, nullptr);
    // t=2: x=[obs2 | emb], h:h1 -> h0
    gru_cell_wmma<<<grid, block, 0, stream>>>(obs_b + 128, OBSLD, FEAT,
                                              emb_b,       HID,   HID,
                                              h1, Wih_b, Whh_b, b_ih, b_hh,
                                              h0, nullptr);
    // pass 4: x=[out0 | obs1], h:h0 -> h1
    gru_cell_wmma<<<grid, block, 0, stream>>>(out0,        HID,   HID,
                                              obs_b + 64,  OBSLD, FEAT,
                                              h0, Wih_b, Whh_b, b_ih, b_hh,
                                              h1, nullptr);
    // pass 5: x=[h1 | obs2], h:h1 -> h0 and f32 -> d_out
    gru_cell_wmma<<<grid, block, 0, stream>>>(h1,          HID,   HID,
                                              obs_b + 128, OBSLD, FEAT,
                                              h1, Wih_b, Whh_b, b_ih, b_hh,
                                              h0, out);
}
